// SingleHeadAttention_86182813762321
// MI455X (gfx1250) — matrix operand order repaired and checked by reading, not compiled, not run on hardware
//
#include <hip/hip_runtime.h>
#include <stdint.h>

#define SEQ   4096
#define EMB   768
#define HD    64
#define BATCH 4

typedef __bf16        v16bf __attribute__((ext_vector_type(16)));
typedef float         v8f   __attribute__((ext_vector_type(8)));
typedef unsigned int  u32x4 __attribute__((ext_vector_type(4)));
typedef unsigned int  u32x2 __attribute__((ext_vector_type(2)));

// Exact pointee type of the async-copy builtin params (from compiler diag).
typedef int async_v4i __attribute__((vector_size(16)));

// Probe the gfx1250 async global->LDS copy builtin (ASYNCcnt path).
#if defined(__has_builtin)
#  if __has_builtin(__builtin_amdgcn_global_load_async_to_lds_b128)
#    define HAVE_ASYNC_LDS 1
#  endif
#endif
#ifndef HAVE_ASYNC_LDS
#  define HAVE_ASYNC_LDS 0
#endif

#if HAVE_ASYNC_LDS
// Copy 16 bytes per lane from global to LDS asynchronously (ASYNCcnt).
__device__ __forceinline__ void async_copy16(const unsigned short* g,
                                             unsigned short* l) {
    void* gv = (void*)g;   // shed const in generic AS first
    void* lv = (void*)l;
    __builtin_amdgcn_global_load_async_to_lds_b128(
        (__attribute__((address_space(1))) async_v4i*)gv,
        (__attribute__((address_space(3))) async_v4i*)lv,
        0, 0);
}
#endif

// Native f32 -> bf16 convert (RNE).
__device__ __forceinline__ unsigned short f32_bf16(float f) {
    __bf16 h = (__bf16)f;
    return __builtin_bit_cast(unsigned short, h);
}

union FragCvt {
    u32x4 q[2];
    v16bf bf;
};

// Load a 16-half (32-byte) WMMA fragment as two b128 loads.
__device__ __forceinline__ v16bf load_frag(const unsigned short* p0,
                                           const unsigned short* p1) {
    FragCvt f;
    f.q[0] = *(const u32x4*)p0;
    f.q[1] = *(const u32x4*)p1;
    return f.bf;
}

__device__ __forceinline__ v8f zero8() {
    v8f z = {0.f, 0.f, 0.f, 0.f, 0.f, 0.f, 0.f, 0.f};
    return z;
}

// ---------------------------------------------------------------------------
// Kernel 1: fused QKV projection.  X(16384x768) @ W(768x64) + b, x3 -> bf16.
// Block = 256 threads (8 waves), 128 output rows.  Q is pre-scaled by
// 1/sqrt(64) so the attention hot loop needs no score scaling.
// ---------------------------------------------------------------------------
__global__ void __launch_bounds__(256, 2)
qkv_proj_kernel(const float* __restrict__ X,
                const float* __restrict__ Wq, const float* __restrict__ bq,
                const float* __restrict__ Wk, const float* __restrict__ bk,
                const float* __restrict__ Wv, const float* __restrict__ bv,
                unsigned short* __restrict__ Qb,
                unsigned short* __restrict__ Kb,
                unsigned short* __restrict__ Vb)
{
    __shared__ unsigned short Xs[128 * 32];        // A tile, bf16, row-major
    __shared__ unsigned short Wt[3][64 * 32];      // W^T tile per matrix: [n][k]

    const int tid  = threadIdx.x;
    const int wid  = tid >> 5;
    const int lane = tid & 31;
    const int g    = lane >> 4;     // half-wave group
    const int ln   = lane & 15;
    const int rowBase = blockIdx.x * 128;

    const float* Wm[3] = {Wq, Wk, Wv};
    const float* Bm[3] = {bq, bk, bv};
    unsigned short* Om[3] = {Qb, Kb, Vb};

    v8f acc[3][4];
    #pragma unroll
    for (int m = 0; m < 3; ++m)
        #pragma unroll
        for (int nt = 0; nt < 4; ++nt)
            acc[m][nt] = zero8();

    for (int k0 = 0; k0 < EMB; k0 += 32) {
        // stage X tile (128x32 f32 -> bf16), b128 global loads
        #pragma unroll
        for (int i = 0; i < 4; ++i) {
            int e = (i * 256 + tid) * 4;           // 0..4092, step 4
            int r = e >> 5, c = e & 31;
            float4 x = *(const float4*)&X[(size_t)(rowBase + r) * EMB + k0 + c];
            union { unsigned short us[4]; u32x2 v; } pk;
            pk.us[0] = f32_bf16(x.x);
            pk.us[1] = f32_bf16(x.y);
            pk.us[2] = f32_bf16(x.z);
            pk.us[3] = f32_bf16(x.w);
            *(u32x2*)&Xs[e] = pk.v;
        }
        // stage 3 weight slices transposed (32x64 f32 -> bf16 [n][k])
        #pragma unroll
        for (int m = 0; m < 3; ++m) {
            #pragma unroll
            for (int i = 0; i < 2; ++i) {
                int e = (i * 256 + tid) * 4;       // 0..2044, step 4
                int kk = e >> 6, n = e & 63;
                float4 w = *(const float4*)&Wm[m][(size_t)(k0 + kk) * HD + n];
                Wt[m][(n + 0) * 32 + kk] = f32_bf16(w.x);
                Wt[m][(n + 1) * 32 + kk] = f32_bf16(w.y);
                Wt[m][(n + 2) * 32 + kk] = f32_bf16(w.z);
                Wt[m][(n + 3) * 32 + kk] = f32_bf16(w.w);
            }
        }
        __syncthreads();

        // A fragment: 16x32 bf16.  lanes 0-15: K 0-7 / 16-23; lanes 16-31: +8.
        const int arow = wid * 16 + ln;
        v16bf a = load_frag(&Xs[arow * 32 + g * 8],
                            &Xs[arow * 32 + 16 + g * 8]);
        #pragma unroll
        for (int m = 0; m < 3; ++m) {
            #pragma unroll
            for (int nt = 0; nt < 4; ++nt) {
                // B fragment 32x16: lane = n, k 0-7 / 16-23 per half-wave (+8 for upper)
                const unsigned short* bp = &Wt[m][(nt * 16 + ln) * 32 + g * 8];
                v16bf bfrag = load_frag(bp, bp + 16);
                acc[m][nt] = __builtin_amdgcn_wmma_f32_16x16x32_bf16(
                    false, a, false, bfrag, (short)0, acc[m][nt], false, false);
            }
        }
        __syncthreads();
    }

    // epilogue: bias (+ 1/sqrt(d) scale for Q), bf16 store row-major [row][64]
    #pragma unroll
    for (int m = 0; m < 3; ++m) {
        #pragma unroll
        for (int nt = 0; nt < 4; ++nt) {
            float bias = Bm[m][nt * 16 + ln];
            #pragma unroll
            for (int r = 0; r < 8; ++r) {
                int row = rowBase + wid * 16 + r + g * 8;   // C: reg=row, lane=col
                float v = acc[m][nt][r] + bias;
                if (m == 0) v *= 0.125f;                    // fold 1/sqrt(64) into Q
                Om[m][(size_t)row * HD + nt * 16 + ln] = f32_bf16(v);
            }
        }
    }
}

// ---------------------------------------------------------------------------
// Kernel 2: causal flash attention over bf16 Q/K/V, f32 accumulate/output.
// Block = 128 threads (4 waves) x 64 query rows -> 256 blocks for balance.
// K tile staged via async global->LDS copy (contiguous 8KB, layout unchanged);
// V tile staged manually (transposed).  The j-loop is split into a fully
// unmasked part (uniform count per wave) and one masked diagonal tile.
// ---------------------------------------------------------------------------
__global__ void __launch_bounds__(128, 4)
flash_attn_kernel(const unsigned short* __restrict__ Qb,
                  const unsigned short* __restrict__ Kb,
                  const unsigned short* __restrict__ Vb,
                  float* __restrict__ Out)
{
    __shared__ unsigned short Ks[64 * 64];          // K tile: [key][d] (as in mem)
    __shared__ unsigned short Vt[64 * 64];          // V^T tile: [d][key]
    __shared__ unsigned short Ps[4 * 16 * 64];      // per-wave P (C->A relayout)

    const int tid  = threadIdx.x;
    const int wid  = tid >> 5;
    const int lane = tid & 31;
    const int g    = lane >> 4;
    const int ln   = lane & 15;
    const int b    = blockIdx.y;
    const int qblk = blockIdx.x;                    // 0..63
    const int qbase = qblk * 64 + wid * 16;         // this wave's first query row
    const size_t batchOff = (size_t)b * SEQ;

    const float NEG_INF = -__builtin_inff();

    // Q A-fragments (pre-scaled by 1/sqrt(d)): two 16x32 tiles, loaded once
    v16bf aq[2];
    {
        const unsigned short* qrow = Qb + (batchOff + qbase + ln) * HD;
        #pragma unroll
        for (int f = 0; f < 2; ++f)
            aq[f] = load_frag(qrow + f * 32 + g * 8,
                              qrow + f * 32 + 16 + g * 8);
    }

    v8f o[4];
    #pragma unroll
    for (int nt = 0; nt < 4; ++nt) o[nt] = zero8();
    float m[8], l[8];
    #pragma unroll
    for (int r = 0; r < 8; ++r) { m[r] = NEG_INF; l[r] = 0.0f; }

    auto process_tile = [&](int jbase, bool domask) {
        // --- K tile: contiguous 8KB, async copy global -> LDS (ASYNCcnt) ---
        {
            const unsigned short* gsrc = Kb + (batchOff + jbase) * HD;
            const int off = tid * 8;                // 16 bytes per lane
            #pragma unroll
            for (int c = 0; c < 4; ++c) {
#if HAVE_ASYNC_LDS
                async_copy16(gsrc + off + c * 1024, Ks + off + c * 1024);
#else
                *(u32x4*)&Ks[off + c * 1024] = *(const u32x4*)&gsrc[off + c * 1024];
#endif
            }
        }
        // --- V tile transposed (overlaps with the async K copy) ---
        #pragma unroll
        for (int i = 0; i < 4; ++i) {
            int e = (i * 128 + tid) * 8;            // 0..4088, step 8
            int key = e >> 6, d0 = e & 63;
            union { unsigned short us[8]; u32x4 v; } vv;
            vv.v = *(const u32x4*)&Vb[(batchOff + jbase + key) * HD + d0];
            #pragma unroll
            for (int d = 0; d < 8; ++d)
                Vt[(d0 + d) * 64 + key] = vv.us[d];
        }
#if HAVE_ASYNC_LDS
        asm volatile("s_wait_asynccnt 0" ::: "memory");
#endif
        __syncthreads();

        // S = Q K^T : 4 n-tiles x 2 k-steps (K tile is B-fragment friendly)
        v8f s[4];
        #pragma unroll
        for (int nt = 0; nt < 4; ++nt) {
            s[nt] = zero8();
            const unsigned short* krow = &Ks[(nt * 16 + ln) * 64];
            #pragma unroll
            for (int f = 0; f < 2; ++f) {
                v16bf bk = load_frag(krow + f * 32 + g * 8,
                                     krow + f * 32 + g * 8 + 16);
                s[nt] = __builtin_amdgcn_wmma_f32_16x16x32_bf16(
                    false, aq[f], false, bk, (short)0, s[nt], false, false);
            }
        }

        // online softmax; every tile has >=1 unmasked key per row, so the
        // running max is finite after this update (no NaN guards needed).
        #pragma unroll
        for (int r = 0; r < 8; ++r) {
            const int qrow = qbase + r + g * 8;     // C layout row for this lane
            float rowmax = NEG_INF;
            #pragma unroll
            for (int nt = 0; nt < 4; ++nt) {
                float v = s[nt][r];
                if (domask) {
                    int key = jbase + nt * 16 + ln;
                    v = (key > qrow) ? NEG_INF : v;
                    s[nt][r] = v;
                }
                rowmax = fmaxf(rowmax, v);
            }
            rowmax = fmaxf(rowmax, __shfl_xor(rowmax, 1));
            rowmax = fmaxf(rowmax, __shfl_xor(rowmax, 2));
            rowmax = fmaxf(rowmax, __shfl_xor(rowmax, 4));
            rowmax = fmaxf(rowmax, __shfl_xor(rowmax, 8));
            float mnew  = fmaxf(m[r], rowmax);      // finite
            float alpha = __expf(m[r] - mnew);      // exp(-inf)=0 on first tile
            float psum  = 0.0f;
            #pragma unroll
            for (int nt = 0; nt < 4; ++nt) {
                float p = __expf(s[nt][r] - mnew);
                psum += p;
                Ps[wid * 1024 + (r + g * 8) * 64 + nt * 16 + ln] = f32_bf16(p);
            }
            psum += __shfl_xor(psum, 1);
            psum += __shfl_xor(psum, 2);
            psum += __shfl_xor(psum, 4);
            psum += __shfl_xor(psum, 8);
            l[r] = l[r] * alpha + psum;
            m[r] = mnew;
            #pragma unroll
            for (int nt = 0; nt < 4; ++nt) o[nt][r] *= alpha;
        }

        // O += P @ V  (P from per-wave LDS as A-frags, V^T tile as B-frags)
        #pragma unroll
        for (int f = 0; f < 2; ++f) {
            const unsigned short* prow = &Ps[wid * 1024 + ln * 64 + f * 32];
            v16bf ap = load_frag(prow + g * 8, prow + 16 + g * 8);
            #pragma unroll
            for (int nt = 0; nt < 4; ++nt) {
                const unsigned short* vrow =
                    &Vt[(nt * 16 + ln) * 64 + f * 32 + g * 8];
                v16bf bv = load_frag(vrow, vrow + 16);
                o[nt] = __builtin_amdgcn_wmma_f32_16x16x32_bf16(
                    false, ap, false, bv, (short)0, o[nt], false, false);
            }
        }
        __syncthreads();
    };

    // Tiles strictly below the diagonal (count is wave-uniform), then the
    // single masked diagonal tile.
    for (int t = 0; t < qblk; ++t) process_tile(t * 64, false);
    process_tile(qblk * 64, true);

    // epilogue: O / l, f32 out (B,S,D)
    #pragma unroll
    for (int r = 0; r < 8; ++r) {
        float inv = (l[r] > 0.0f) ? 1.0f / l[r] : 0.0f;
        int qrow = qbase + r + g * 8;
        #pragma unroll
        for (int nt = 0; nt < 4; ++nt)
            Out[(batchOff + qrow) * HD + nt * 16 + ln] = o[nt][r] * inv;
    }
}

// ---------------------------------------------------------------------------
extern "C" void kernel_launch(void* const* d_in, const int* in_sizes, int n_in,
                              void* d_out, int out_size, void* d_ws, size_t ws_size,
                              hipStream_t stream) {
    const float* X  = (const float*)d_in[0];
    const float* Wq = (const float*)d_in[1];
    const float* bq = (const float*)d_in[2];
    const float* Wk = (const float*)d_in[3];
    const float* bk = (const float*)d_in[4];
    const float* Wv = (const float*)d_in[5];
    const float* bv = (const float*)d_in[6];
    float* Out = (float*)d_out;

    // Workspace: Q/K/V in bf16, 2 MB each (6 MB total)
    unsigned short* Qb = (unsigned short*)d_ws;
    unsigned short* Kb = Qb + (size_t)BATCH * SEQ * HD;
    unsigned short* Vb = Kb + (size_t)BATCH * SEQ * HD;

    qkv_proj_kernel<<<dim3(BATCH * SEQ / 128), dim3(256), 0, stream>>>(
        X, Wq, bq, Wk, bk, Wv, bv, Qb, Kb, Vb);
    flash_attn_kernel<<<dim3(SEQ / 64, BATCH), dim3(128), 0, stream>>>(
        Qb, Kb, Vb, Out);
}
